// HTTNBaseHead_4389456576871
// MI455X (gfx1250) — matrix operand order repaired and checked by reading, not compiled, not run on hardware
//
#include <hip/hip_runtime.h>
#include <hip/hip_bf16.h>

#define DEVFN __device__ __forceinline__

typedef __attribute__((ext_vector_type(16))) __bf16 v16bf;
typedef __attribute__((ext_vector_type(8)))  float  v8f;
typedef __attribute__((ext_vector_type(4)))  unsigned int v4u;

constexpr int Bn  = 4096;   // batch
constexpr int Cn  = 8192;   // classes
constexpr int Dn  = 256;    // feature dim
constexpr int HIDn= 512;    // hidden
constexpr int Hn  = 512;    // heads
constexpr float EMA = 0.99f;

// ---------- workspace layout (bytes) ----------
constexpr size_t OFF_P_F32  = 0;                                   // C*D f32
constexpr size_t OFF_P_BF   = OFF_P_F32  + (size_t)Cn*Dn*4;        // C*D bf16
constexpr size_t OFF_PHEAD  = OFF_P_BF   + (size_t)Cn*Dn*2;        // H*D bf16 (row-major)
constexpr size_t OFF_PHEADT = OFF_PHEAD  + (size_t)Hn*Dn*2;        // D*H bf16 (transposed)
constexpr size_t OFF_PEFF   = OFF_PHEADT + (size_t)Dn*Hn*2;        // C*D bf16
constexpr size_t OFF_HB     = OFF_PEFF   + (size_t)Cn*Dn*2;        // C*HID bf16
constexpr size_t OFF_WB     = OFF_HB     + (size_t)Cn*HIDn*2;      // C*D bf16
constexpr size_t OFF_BIAS   = OFF_WB     + (size_t)Cn*Dn*2;        // C f32
constexpr size_t OFF_ZB     = OFF_BIAS   + (size_t)Cn*4;           // B*D bf16 (row-major)
constexpr size_t OFF_ZT     = OFF_ZB     + (size_t)Bn*Dn*2;        // D*B bf16 (transposed)
constexpr size_t OFF_W1B    = OFF_ZT     + (size_t)Dn*Bn*2;        // HID*D bf16
constexpr size_t OFF_W2B    = OFF_W1B    + (size_t)HIDn*Dn*2;      // (D+1)*HID bf16

union FragU { unsigned int u[8]; v4u q[2]; v16bf v; };

// native bf16 convert (single v_cvt on gfx1250, RNE)
DEVFN unsigned short f2bf(float x) {
  __bf16 h = (__bf16)x;
  return __builtin_bit_cast(unsigned short, h);
}

DEVFN v8f vzero() { v8f v; for (int i = 0; i < 8; ++i) v[i] = 0.f; return v; }

DEVFN v8f wmma_bf(v16bf a, v16bf b, v8f c) {
  return __builtin_amdgcn_wmma_f32_16x16x32_bf16(false, a, false, b, (short)0, c, false, false);
}

// A fragment: 16(M) x 32(K) bf16, row-major source (lda elements; rows 16B-aligned).
// Lane L: m = L&15, g = L>>4. VGPRs 0..7: K = g*16 .. g*16+15
DEVFN v16bf a_frag(const unsigned short* A, int lda) {
  int lane = threadIdx.x & 31;
  int m = lane & 15, g = lane >> 4;
  const unsigned short* row = A + (size_t)m * lda + g * 16;
  FragU x;
  x.q[0] = *(const v4u*)(row);
  x.q[1] = *(const v4u*)(row + 8);
  return x.v;
}

// B fragment from transposed source: B[k][n] = src[n*ldb + k]  (K-pairs contiguous)
// Lane L: n = L&15, g = L>>4. VGPR i: K = g*16 + 2i, +1  -> 32 contiguous bytes
DEVFN v16bf b_frag_T(const unsigned short* src, int ldb, int k0) {
  int lane = threadIdx.x & 31;
  int n = lane & 15, g = lane >> 4;
  const unsigned short* row = src + (size_t)n * ldb + k0 + g * 16;
  FragU x;
  x.q[0] = *(const v4u*)(row);
  x.q[1] = *(const v4u*)(row + 8);
  return x.v;
}

// Same, but rows >= nlim supply zeros (for the D+1 = 257 padded operand)
DEVFN v16bf b_frag_T_guard(const unsigned short* src, int ldb, int k0, int n0, int nlim) {
  int lane = threadIdx.x & 31;
  int n = n0 + (lane & 15), g = lane >> 4;
  FragU x;
  if (n < nlim) {
    const unsigned short* row = src + (size_t)n * ldb + k0 + g * 16;
    x.q[0] = *(const v4u*)(row);
    x.q[1] = *(const v4u*)(row + 8);
  } else {
#pragma unroll
    for (int i = 0; i < 8; ++i) x.u[i] = 0u;
  }
  return x.v;
}

// ---------------- K0: f32 -> bf16 conversions (+ z transpose) ----------------
__global__ __launch_bounds__(256) void k_prep(const float* z, const float* w1, const float* w2,
                                              unsigned short* zb, unsigned short* zT,
                                              unsigned short* w1b, unsigned short* w2b) {
  const int n0 = Bn * Dn, n1 = HIDn * Dn, n2 = (Dn + 1) * HIDn;
  for (int i = blockIdx.x * 256 + threadIdx.x; i < n0 + n1 + n2; i += gridDim.x * 256) {
    if (i < n0) {
      int b = i >> 8, d = i & 255;
      unsigned short v = f2bf(z[i]);
      zb[i] = v;
      zT[(size_t)d * Bn + b] = v;
    } else if (i < n0 + n1) {
      w1b[i - n0] = f2bf(w1[i - n0]);
    } else {
      w2b[i - n0 - n1] = f2bf(w2[i - n0 - n1]);
    }
  }
}

// ---------------- K1: sums = mask.T @ z ; counts ; EMA select -> P ----------------
__global__ __launch_bounds__(256) void k_protos(const float* labels, const unsigned short* zT,
                                                const float* protos, const unsigned char* inited,
                                                float* P_f32, unsigned short* P_bf) {
  __shared__ unsigned short maskA[16 * 32];    // [m=c][k=b], rows 64B apart
  __shared__ float cntp[256];
  __shared__ float counts16[16];

  const int tid = threadIdx.x, wave = tid >> 5;
  const int c0 = blockIdx.x * 16;
  const int j = tid & 15, i = tid >> 4;        // j: c-col, i: b-row (and i+16)

  v8f acc[2] = { vzero(), vzero() };
  float cnt = 0.f;

  for (int b0 = 0; b0 < Bn; b0 += 32) {
    __syncthreads();
    // stage mask tile (transposed labels, >0.5 -> 1.0bf16)
    float l0 = labels[(size_t)(b0 + i) * Cn + (c0 + j)];
    float l1 = labels[(size_t)(b0 + i + 16) * Cn + (c0 + j)];
    unsigned short m0 = (l0 > 0.5f) ? (unsigned short)0x3F80 : 0;
    unsigned short m1 = (l1 > 0.5f) ? (unsigned short)0x3F80 : 0;
    maskA[j * 32 + i]      = m0;
    maskA[j * 32 + i + 16] = m1;
    cnt += (m0 ? 1.f : 0.f) + (m1 ? 1.f : 0.f);
    __syncthreads();
    v16bf a = a_frag(maskA, 32);
#pragma unroll
    for (int p = 0; p < 2; ++p) {
      int n0 = (2 * wave + p) * 16;                  // d-tile
      v16bf b = b_frag_T(zT + (size_t)n0 * Bn, Bn, b0);
      acc[p] = wmma_bf(a, b, acc[p]);
    }
  }

  cntp[tid] = cnt;
  __syncthreads();
  if (tid < 16) {
    float s = 0.f;
    for (int k = 0; k < 16; ++k) s += cntp[k * 16 + tid];
    counts16[tid] = s;
  }
  __syncthreads();

  const int lane = tid & 31, l = lane & 15, g = lane >> 4;
#pragma unroll
  for (int p = 0; p < 2; ++p) {
    int d = (2 * wave + p) * 16 + l;
#pragma unroll
    for (int r = 0; r < 8; ++r) {
      int m = r + 8 * g;
      int c = c0 + m;
      float cc = counts16[m];
      float mean = acc[p][r] / fmaxf(cc, 1.f);
      float pr = protos[(size_t)c * Dn + d];
      float out;
      if (cc > 0.f) out = inited[c] ? (EMA * pr + (1.f - EMA) * mean) : mean;
      else          out = pr;
      P_f32[(size_t)c * Dn + d] = out;
      P_bf[(size_t)c * Dn + d]  = f2bf(out);
    }
  }
}

// ---------------- K2: P_head = P[head_ids] (bf16, row-major + transposed) ----------------
__global__ __launch_bounds__(256) void k_head(const unsigned short* P_bf, const long long* hid,
                                              unsigned short* Phead, unsigned short* PheadT) {
  int idx = blockIdx.x * 256 + threadIdx.x;           // H*D = 131072 threads
  if (idx < Hn * Dn) {
    int h = idx >> 8, d = idx & 255;
    unsigned short v = P_bf[(size_t)hid[h] * Dn + d];
    Phead[idx] = v;
    PheadT[(size_t)d * Hn + h] = v;
  }
}

// ---------------- K3: scores -> softmax -> P_eff = P + alpha @ P_head ----------------
__global__ __launch_bounds__(256) void k_attn(const unsigned short* P_bf, const float* P_f32,
                                              const unsigned short* Phead, const unsigned short* PheadT,
                                              unsigned short* Peff) {
  __shared__ float sc[16 * 512];                 // scores f32 (32KB)
  __shared__ unsigned short al[16 * 512];        // alpha bf16 (16KB)
  __shared__ float red[256];
  __shared__ float rowv[16];

  const int tid = threadIdx.x, wave = tid >> 5;
  const int c0 = blockIdx.x * 16;
  const int lane = tid & 31, l = lane & 15, g = lane >> 4;

  // ---- phase 1: scores = P @ P_head.T / 16 ----
  v8f acc[4] = { vzero(), vzero(), vzero(), vzero() };
#pragma unroll
  for (int ks = 0; ks < Dn / 32; ++ks) {
    v16bf a = a_frag(P_bf + (size_t)c0 * Dn + ks * 32, Dn);
#pragma unroll
    for (int tt = 0; tt < 4; ++tt) {
      int n0 = (wave + 8 * tt) * 16;
      v16bf b = b_frag_T(Phead + (size_t)n0 * Dn, Dn, ks * 32);
      acc[tt] = wmma_bf(a, b, acc[tt]);
    }
  }
#pragma unroll
  for (int tt = 0; tt < 4; ++tt) {
    int n0 = (wave + 8 * tt) * 16;
#pragma unroll
    for (int r = 0; r < 8; ++r) {
      int m = r + 8 * g;
      sc[m * 512 + n0 + l] = acc[tt][r] * 0.0625f;   // 1/sqrt(256)
    }
  }
  __syncthreads();

  // ---- softmax over H=512 (row = tid>>4 handles 32 strided cols) ----
  {
    int row = tid >> 4, l16 = tid & 15;
    float s[32], mx = -3.402823e38f;
#pragma unroll
    for (int jj = 0; jj < 32; ++jj) { s[jj] = sc[row * 512 + l16 + 16 * jj]; mx = fmaxf(mx, s[jj]); }
    red[tid] = mx;
    __syncthreads();
    if (l16 == 0) {
      float m2 = red[row * 16];
      for (int k = 1; k < 16; ++k) m2 = fmaxf(m2, red[row * 16 + k]);
      rowv[row] = m2;
    }
    __syncthreads();
    float rm = rowv[row], sum = 0.f, e[32];
#pragma unroll
    for (int jj = 0; jj < 32; ++jj) { e[jj] = __expf(s[jj] - rm); sum += e[jj]; }
    __syncthreads();
    red[tid] = sum;
    __syncthreads();
    if (l16 == 0) {
      float t = 0.f;
      for (int k = 0; k < 16; ++k) t += red[row * 16 + k];
      rowv[row] = t;
    }
    __syncthreads();
    float inv = 1.f / rowv[row];
#pragma unroll
    for (int jj = 0; jj < 32; ++jj) al[row * 512 + l16 + 16 * jj] = f2bf(e[jj] * inv);
  }
  __syncthreads();

  // ---- phase 2: P_eff = P + alpha @ P_head  (B via PheadT, direct b128 gathers) ----
  v8f acc2[2] = { vzero(), vzero() };
#pragma unroll
  for (int ks = 0; ks < Hn / 32; ++ks) {
    v16bf a = a_frag(al + ks * 32, 512);
#pragma unroll
    for (int p = 0; p < 2; ++p) {
      int n0 = (2 * wave + p) * 16;                  // d-tile
      v16bf b = b_frag_T(PheadT + (size_t)n0 * Hn, Hn, ks * 32);
      acc2[p] = wmma_bf(a, b, acc2[p]);
    }
  }
#pragma unroll
  for (int p = 0; p < 2; ++p) {
    int d = (2 * wave + p) * 16 + l;
#pragma unroll
    for (int r = 0; r < 8; ++r) {
      int m = r + 8 * g, c = c0 + m;
      float pe = acc2[p][r] + P_f32[(size_t)c * Dn + d];
      Peff[(size_t)c * Dn + d] = f2bf(pe);
    }
  }
}

// ---------------- K4: h = relu(P_eff @ w1.T + b1) ----------------
__global__ __launch_bounds__(256) void k_hidden(const unsigned short* Peff, const unsigned short* w1b,
                                                const float* b1, unsigned short* hb) {
  const int tid = threadIdx.x, wave = tid >> 5;
  const int c0 = blockIdx.x * 16;
  const int lane = tid & 31, l = lane & 15, g = lane >> 4;
  v8f acc[4] = { vzero(), vzero(), vzero(), vzero() };
#pragma unroll
  for (int ks = 0; ks < Dn / 32; ++ks) {
    v16bf a = a_frag(Peff + (size_t)c0 * Dn + ks * 32, Dn);
#pragma unroll
    for (int tt = 0; tt < 4; ++tt) {
      int n0 = (wave + 8 * tt) * 16;
      v16bf b = b_frag_T(w1b + (size_t)n0 * Dn, Dn, ks * 32);
      acc[tt] = wmma_bf(a, b, acc[tt]);
    }
  }
#pragma unroll
  for (int tt = 0; tt < 4; ++tt) {
    int col = (wave + 8 * tt) * 16 + l;
    float bias = b1[col];
#pragma unroll
    for (int r = 0; r < 8; ++r) {
      int m = r + 8 * g;
      float v = fmaxf(acc[tt][r] + bias, 0.f);
      hb[(size_t)(c0 + m) * HIDn + col] = f2bf(v);
    }
  }
}

// ---------------- K5: wb = h @ w2.T + b2 ; split W (bf16) / bias (f32) ----------------
__global__ __launch_bounds__(256) void k_wb(const unsigned short* hb, const unsigned short* w2b,
                                            const float* b2, unsigned short* Wb, float* biasv) {
  const int tid = threadIdx.x, wave = tid >> 5;
  const int c0 = blockIdx.x * 16;
  const int lane = tid & 31, l = lane & 15, g = lane >> 4;
  v8f acc[3] = { vzero(), vzero(), vzero() };      // 17 tiles over 8 waves
#pragma unroll
  for (int ks = 0; ks < HIDn / 32; ++ks) {
    v16bf a = a_frag(hb + (size_t)c0 * HIDn + ks * 32, HIDn);
#pragma unroll
    for (int ti = 0; ti < 3; ++ti) {
      int t = wave + 8 * ti;
      if (t < 17) {
        v16bf b = b_frag_T_guard(w2b, HIDn, ks * 32, t * 16, Dn + 1);
        acc[ti] = wmma_bf(a, b, acc[ti]);
      }
    }
  }
#pragma unroll
  for (int ti = 0; ti < 3; ++ti) {
    int t = wave + 8 * ti;
    if (t < 17) {
      int col = t * 16 + l;
#pragma unroll
      for (int r = 0; r < 8; ++r) {
        int m = r + 8 * g, c = c0 + m;
        if (col < Dn) {
          Wb[(size_t)c * Dn + col] = f2bf(acc[ti][r] + b2[col]);
        } else if (col == Dn) {
          biasv[c] = acc[ti][r] + b2[Dn];
        }
      }
    }
  }
}

// ---------------- K6: logits = z @ W.T + b  (32 B-rows x 256 C-cols per block) ----------------
__global__ __launch_bounds__(256) void k_logits(const unsigned short* zb, const unsigned short* Wb,
                                                const float* biasv, float* out) {
  const int tid = threadIdx.x, wave = tid >> 5;
  const int b0 = blockIdx.x * 32;
  const int cblk = blockIdx.y * 256;
  const int lane = tid & 31, l = lane & 15, g = lane >> 4;
  v8f acc[2][2] = { { vzero(), vzero() }, { vzero(), vzero() } };   // [m-tile][n-tile]
#pragma unroll
  for (int ks = 0; ks < Dn / 32; ++ks) {
    v16bf a0 = a_frag(zb + (size_t)b0 * Dn + ks * 32, Dn);
    v16bf a1 = a_frag(zb + (size_t)(b0 + 16) * Dn + ks * 32, Dn);
#pragma unroll
    for (int p = 0; p < 2; ++p) {
      int n0 = (2 * wave + p) * 16;
      v16bf b = b_frag_T(Wb + (size_t)(cblk + n0) * Dn, Dn, ks * 32);
      acc[0][p] = wmma_bf(a0, b, acc[0][p]);
      acc[1][p] = wmma_bf(a1, b, acc[1][p]);
    }
  }
#pragma unroll
  for (int p = 0; p < 2; ++p) {
    int c = cblk + (2 * wave + p) * 16 + l;
    float bias = biasv[c];
#pragma unroll
    for (int mi = 0; mi < 2; ++mi) {
#pragma unroll
      for (int r = 0; r < 8; ++r) {
        int m = r + 8 * g;
        out[(size_t)(b0 + mi * 16 + m) * Cn + c] = acc[mi][p][r] + bias;
      }
    }
  }
}

extern "C" void kernel_launch(void* const* d_in, const int* in_sizes, int n_in,
                              void* d_out, int out_size, void* d_ws, size_t ws_size,
                              hipStream_t stream) {
  const float*         z      = (const float*)d_in[0];
  const float*         labels = (const float*)d_in[1];
  const float*         protos = (const float*)d_in[2];
  const unsigned char* inited = (const unsigned char*)d_in[3];
  const long long*     hid    = (const long long*)d_in[4];
  const float*         w1     = (const float*)d_in[5];
  const float*         b1     = (const float*)d_in[6];
  const float*         w2     = (const float*)d_in[7];
  const float*         b2     = (const float*)d_in[8];

  char* ws = (char*)d_ws;
  float*          P_f32  = (float*)(ws + OFF_P_F32);
  unsigned short* P_bf   = (unsigned short*)(ws + OFF_P_BF);
  unsigned short* Phead  = (unsigned short*)(ws + OFF_PHEAD);
  unsigned short* PheadT = (unsigned short*)(ws + OFF_PHEADT);
  unsigned short* Peff   = (unsigned short*)(ws + OFF_PEFF);
  unsigned short* hb     = (unsigned short*)(ws + OFF_HB);
  unsigned short* Wb     = (unsigned short*)(ws + OFF_WB);
  float*          biasv  = (float*)(ws + OFF_BIAS);
  unsigned short* zb     = (unsigned short*)(ws + OFF_ZB);
  unsigned short* zT     = (unsigned short*)(ws + OFF_ZT);
  unsigned short* w1b    = (unsigned short*)(ws + OFF_W1B);
  unsigned short* w2b    = (unsigned short*)(ws + OFF_W2B);

  k_prep  <<<dim3(5122), dim3(256), 0, stream>>>(z, w1, w2, zb, zT, w1b, w2b);
  k_protos<<<dim3(Cn / 16), dim3(256), 0, stream>>>(labels, zT, protos, inited, P_f32, P_bf);
  k_head  <<<dim3((Hn * Dn) / 256), dim3(256), 0, stream>>>(P_bf, hid, Phead, PheadT);
  k_attn  <<<dim3(Cn / 16), dim3(256), 0, stream>>>(P_bf, P_f32, Phead, PheadT, Peff);
  k_hidden<<<dim3(Cn / 16), dim3(256), 0, stream>>>(Peff, w1b, b1, hb);
  k_wb    <<<dim3(Cn / 16), dim3(256), 0, stream>>>(hb, w2b, b2, Wb, biasv);
  k_logits<<<dim3(Bn / 32, Cn / 256), dim3(256), 0, stream>>>(zb, Wb, biasv, (float*)d_out);
}
